// MultiLevelModel_87428354277623
// MI455X (gfx1250) — matrix operand order repaired and checked by reading, not compiled, not run on hardware
//
#include <hip/hip_runtime.h>

typedef __attribute__((ext_vector_type(16))) _Float16 v16h;
typedef __attribute__((ext_vector_type(8)))  _Float16 v8h;
typedef __attribute__((ext_vector_type(8)))  float    v8f;
typedef __attribute__((ext_vector_type(4)))  unsigned int v4u;

#define B_    4
#define C_    128
#define H_    128
#define W_    128
#define HW_   (H_ * W_)
#define PADP  4                 // halo = dilation * (K-1)/2 = 4
#define HP    (H_ + 2 * PADP)   // 136
#define WP    (W_ + 2 * PADP)   // 136
#define NCOL  24                // neighbor columns for a 16-wide tile: 16 + 2*4
#define NQ    120               // 5 rows * 24 cols of neighbor pixels
#define NTILE 8                 // ceil(120/16)

// ---------------------------------------------------------------------------
// Kernel 1: one block per (b,h) image row.
//  - coalesced f32 reads of feat [B,C,H,W]
//  - LDS-staged transpose -> zero-padded f16 feat16p [B,HP,WP,C] (contiguous
//    channels per pixel; halo pre-zeroed by hipMemsetAsync)
//  - folds the 1x1 classifier: s_pad[b,h+4,w+4] = w_cls . feat[:, h, w]
// LDS pitch 130 halves (65 dwords): write banks (w+c/2) and read banks
// (w + 4*c8 + j) are conflict-free mod 64.
// ---------------------------------------------------------------------------
__launch_bounds__(256)
__global__ void prep_kernel(const float* __restrict__ feat,
                            const float* __restrict__ w_cls,
                            _Float16* __restrict__ feat16p,
                            float* __restrict__ s_pad) {
  __shared__ _Float16 T[W_][130];
  __shared__ float partial[256];

  const int b = blockIdx.x / H_;
  const int h = blockIdx.x - b * H_;
  const int wl   = threadIdx.x & (W_ - 1);   // w
  const int cpar = threadIdx.x >> 7;         // channel parity (0/1)

  const float* src = feat + (size_t)b * C_ * HW_ + (size_t)h * W_;
  float acc = 0.0f;
#pragma unroll 8
  for (int it = 0; it < C_ / 2; ++it) {
    const int c = it * 2 + cpar;
    const float v = src[(size_t)c * HW_ + wl];       // lanes contiguous in w
    acc = fmaf(w_cls[c], v, acc);
    T[wl][c] = (_Float16)v;
  }
  partial[threadIdx.x] = acc;
  __syncthreads();

  if (threadIdx.x < W_) {
    const float sv = partial[threadIdx.x] + partial[threadIdx.x + 128];
    s_pad[(size_t)(b * HP + h + PADP) * WP + threadIdx.x + PADP] = sv;
  }

  // stream out [w][c] as contiguous 16B chunks: elem e = i*256+t,
  // w = e>>4, c8 = e&15  ->  dst + w*128 + c8*8 halves (fully coalesced)
  _Float16* dst =
      feat16p + ((size_t)((b * HP + h + PADP) * WP + PADP)) * C_;
  const unsigned int* tp = (const unsigned int*)&T[0][0];
#pragma unroll
  for (int i = 0; i < 8; ++i) {
    const int e  = i * 256 + threadIdx.x;
    const int w  = e >> 4;
    const int c8 = e & 15;
    const int base = w * 65 + c8 * 4;                // dword index into T
    v4u d;
    d.x = tp[base + 0];
    d.y = tp[base + 1];
    d.z = tp[base + 2];
    d.w = tp[base + 3];
    *(v4u*)(dst + (size_t)w * C_ + c8 * 8) = d;
  }
}

// ---------------------------------------------------------------------------
// Kernel 2: one wave per 16-pixel row segment.
//   Gram G[16x120] = tile_pixels x neighbor_pixels via v_wmma_f32_16x16x32_f16
//   (8 N-tiles x 4 K-chunks, all loads unconditional thanks to the halo),
//   G -> LDS, then per-pixel gather of the 25 dilated-neighborhood entries,
//   relu+softmax, logits = sum attn * s + bias.
// ---------------------------------------------------------------------------
__launch_bounds__(128)
__global__ void attn_kernel(const _Float16* __restrict__ feat16p,
                            const float* __restrict__ s_pad,
                            const float* __restrict__ b_cls,
                            float* __restrict__ out) {
  __shared__ float G[4][NTILE][16][17];              // +1 pad: bank-safe
  const int lane = threadIdx.x & 31;
  const int wv   = threadIdx.x >> 5;
  const int seg  = blockIdx.x * 4 + wv;              // 4096 segments
  const int wt   = seg & 7;
  const int h    = (seg >> 3) & (H_ - 1);
  const int b    = seg >> 10;
  const int w0   = wt * 16;

  // ---- A fragments: 16 tile pixels (M) x 128 channels (K), f16 ----
  // lane<16: row M=lane, K chunks {k0+0..7, k0+16..23};
  // lane>=16: row M=lane-16, K chunks {k0+8..15, k0+24..31}.
  const int am = lane & 15;
  const int ah = (lane >> 4) & 1;
  const _Float16* arow =
      feat16p + ((size_t)((b * HP + h + PADP) * WP + w0 + PADP + am)) * C_;
  v16h a[4];
#pragma unroll
  for (int kc = 0; kc < 4; ++kc) {
    const int k0 = kc * 32 + ah * 8;
    union { v16h v; v8h h2[2]; } u;
    u.h2[0] = *(const v8h*)(arow + k0);
    u.h2[1] = *(const v8h*)(arow + k0 + 16);
    a[kc] = u.v;
  }

  // ---- B fragments: lane owns neighbor column q = 16t + lane%16 ----
  // lane<16 holds K=k0+{0..7,16..23} of its column, lane>=16 holds K=k0+{8..15,24..31}.
  // q in [120,128) is never gathered -> clamp to 119 (halo keeps it legal).
  const int bh = ((lane >> 4) & 1) * 8;
#pragma unroll
  for (int t = 0; t < NTILE; ++t) {
    int q = t * 16 + (lane & 15);
    q = q < NQ ? q : (NQ - 1);
    const int di  = q / NCOL;                        // 0..4
    const int col = q - di * NCOL;                   // 0..23
    // padded coords: hp = h + 2*di, wp = w0 + col  (always in-bounds)
    const _Float16* brow =
        feat16p + ((size_t)((b * HP + h + 2 * di) * WP + w0 + col)) * C_;
    v8f c = {};
#pragma unroll
    for (int kc = 0; kc < 4; ++kc) {
      const int k0 = kc * 32 + bh;
      union { v16h v; v8h h2[2]; } u;
      u.h2[0] = *(const v8h*)(brow + k0);
      u.h2[1] = *(const v8h*)(brow + k0 + 16);
      c = __builtin_amdgcn_wmma_f32_16x16x32_f16(
              false, a[kc], false, u.v, (short)0, c, false, false);
    }
    // C layout: lane<16 -> M=r, N=lane ; lane>=16 -> M=8+r, N=lane-16
    const int n  = lane & 15;
    const int mb = (lane >> 4) * 8;
#pragma unroll
    for (int r = 0; r < 8; ++r) G[wv][t][mb + r][n] = c[r];
  }
  __syncthreads();

  // ---- per-pixel gather + relu + softmax + weighted sum of s ----
  if (lane < 16) {
    const int m = lane;
    float vals[25];
    float mx = 0.0f;                                 // relu floor
#pragma unroll
    for (int di = 0; di < 5; ++di)
#pragma unroll
      for (int dj = 0; dj < 5; ++dj) {
        const int q = di * NCOL + m + 2 * dj;        // <= 119
        float v = G[wv][q >> 4][m][q & 15];
        v = fmaxf(v, 0.0f);                          // clamp(min=0)
        vals[di * 5 + dj] = v;
        mx = fmaxf(mx, v);
      }
    float Z = 0.0f, num = 0.0f;
#pragma unroll
    for (int di = 0; di < 5; ++di) {
      const float* srow = s_pad + (size_t)(b * HP + h + 2 * di) * WP + w0 + m;
#pragma unroll
      for (int dj = 0; dj < 5; ++dj) {
        const float e = __expf(vals[di * 5 + dj] - mx);
        Z += e;                                      // padded: exp(0-mx)
        num = fmaf(e, srow[2 * dj], num);            // padded s = 0
      }
    }
    out[(b * H_ + h) * W_ + w0 + m] = num / Z + b_cls[0];
  }
}

// ---------------------------------------------------------------------------
extern "C" void kernel_launch(void* const* d_in, const int* in_sizes, int n_in,
                              void* d_out, int out_size, void* d_ws,
                              size_t ws_size, hipStream_t stream) {
  const float* feat  = (const float*)d_in[0];   // [4,128,128,128] f32
  const float* w_cls = (const float*)d_in[1];   // [1,128] f32
  const float* b_cls = (const float*)d_in[2];   // [1] f32
  float* out = (float*)d_out;                   // [4,1,128,128] f32

  const size_t feat16_bytes = (size_t)B_ * HP * WP * C_ * sizeof(_Float16);
  const size_t spad_bytes   = (size_t)B_ * HP * WP * sizeof(float);
  _Float16* feat16p = (_Float16*)d_ws;
  float* s_pad = (float*)((char*)d_ws + feat16_bytes);

  // zero halos (and everything else) — graph-capture-legal async memset
  hipMemsetAsync(d_ws, 0, feat16_bytes + spad_bytes, stream);

  prep_kernel<<<B_ * H_, 256, 0, stream>>>(feat, w_cls, feat16p, s_pad);

  const int nseg = B_ * H_ * (W_ / 16);         // 4096 waves
  attn_kernel<<<nseg / 4, 128, 0, stream>>>(feat16p, s_pad, b_cls, out);
}
